// MultiHeadAttention_57166014710202
// MI455X (gfx1250) — matrix operand order repaired and checked by reading, not compiled, not run on hardware
//
#include <hip/hip_runtime.h>
#include <hip/hip_bf16.h>
#include <math.h>

// ---------------------------------------------------------------------------
// MultiHeadAttention forward, bf16 WMMA pipeline for gfx1250 (MI455X)
//   B=2, T=2048, D=1024, H=16, Dh=64
// ---------------------------------------------------------------------------

#define B_   2
#define T_   2048
#define D_   1024
#define H_   16
#define DH_  64
#define BT_  (B_ * T_)          // 4096

typedef __attribute__((ext_vector_type(16))) __bf16 v16bf;
typedef __attribute__((ext_vector_type(8)))  float  v8f;
typedef int v4i_b128 __attribute__((vector_size(16)));   // matches builtin param

#define DEV __device__ __forceinline__

// Async global->LDS (CDNA5): guarded so a missing builtin degrades to the
// synchronous path instead of breaking the build.
#if defined(__AMDGCN__)
#  if __has_builtin(__builtin_amdgcn_global_load_async_to_lds_b128) && \
      __has_builtin(__builtin_amdgcn_s_wait_asynccnt)
#    define USE_ASYNC_LDS 1
#  endif
#endif

#ifdef USE_ASYNC_LDS
// Signature (from hipcc diagnostic): param0 = generic int4* (global source),
// param1 = LDS destination, then imm offset + imm cpol.
#define ASYNC_CP_B128(gsrc, ldst)                                        \
  __builtin_amdgcn_global_load_async_to_lds_b128(                        \
      (v4i_b128*)(gsrc),                                                 \
      (__attribute__((address_space(3))) v4i_b128*)(ldst), 0, 0)
#define ASYNC_WAIT0() __builtin_amdgcn_s_wait_asynccnt(0)
#endif

union FragCvt { uint4 u[2]; v16bf v; unsigned short s[16]; };

DEV unsigned short f2bf(float f) {            // round-to-nearest-even f32->bf16
  unsigned u = __float_as_uint(f);
  u += 0x7FFFu + ((u >> 16) & 1u);
  return (unsigned short)(u >> 16);
}

// A-operand fragment (16x32 bf16): lane row = lane&15,
// k-chunks [c0..c0+7] and [c0+16..c0+23], c0 = kbase + 8*(lane>>4)
DEV v16bf ldA(const unsigned short* rowPtr, int c0) {
  FragCvt f;
  f.u[0] = *reinterpret_cast<const uint4*>(rowPtr + c0);
  f.u[1] = *reinterpret_cast<const uint4*>(rowPtr + c0 + 16);
  return f.v;
}
// B-operand fragment (32x16 bf16): lane col = lane&15,
// k-chunks [c0..c0+7] and [c0+16..c0+23], c0 = kbase + 8*(lane>>4)
DEV v16bf ldB(const unsigned short* rowPtr, int c0) {
  FragCvt f;
  f.u[0] = *reinterpret_cast<const uint4*>(rowPtr + c0);
  f.u[1] = *reinterpret_cast<const uint4*>(rowPtr + c0 + 16);
  return f.v;
}

DEV v8f wmma_bf16(v16bf a, v16bf b, v8f c) {
  return __builtin_amdgcn_wmma_f32_16x16x32_bf16(
      /*neg_a=*/false, a, /*neg_b=*/false, b,
      /*c_mod=*/(short)0, c, /*reuse_a=*/false, /*reuse_b=*/false);
}

DEV v8f zero8() { v8f z = {0.f,0.f,0.f,0.f,0.f,0.f,0.f,0.f}; return z; }

DEV float redmax16(float x) {   // reduce over a 16-lane half (one C-tile row set)
  x = fmaxf(x, __shfl_xor(x, 1));
  x = fmaxf(x, __shfl_xor(x, 2));
  x = fmaxf(x, __shfl_xor(x, 4));
  x = fmaxf(x, __shfl_xor(x, 8));
  return x;
}
DEV float redsum16(float x) {
  x += __shfl_xor(x, 1);
  x += __shfl_xor(x, 2);
  x += __shfl_xor(x, 4);
  x += __shfl_xor(x, 8);
  return x;
}

// ---------------------------------------------------------------------------
// Kernel 1: f32 -> bf16 conversion (vectorized)
// ---------------------------------------------------------------------------
__global__ __launch_bounds__(256)
void mha_cvt_bf16(const float* __restrict__ in, unsigned short* __restrict__ out, int n) {
  int i = (blockIdx.x * 256 + threadIdx.x) * 4;
  if (i + 3 < n) {
    float4 f = *reinterpret_cast<const float4*>(in + i);
    unsigned lo = (unsigned)f2bf(f.x) | ((unsigned)f2bf(f.y) << 16);
    unsigned hi = (unsigned)f2bf(f.z) | ((unsigned)f2bf(f.w) << 16);
    *reinterpret_cast<uint2*>(out + i) = make_uint2(lo, hi);
  } else {
    for (; i < n; ++i) out[i] = f2bf(in[i]);
  }
}

// ---------------------------------------------------------------------------
// Kernel 2: Y = X @ W^T   (X:[4096,1024] bf16, W:[1024,1024] bf16 row-major)
//   mode 0: store bf16 scattered to head-major [B,H,T,Dh]
//   mode 1: store f32 row-major [4096,1024]
// 128x128 tile / workgroup, 8 waves, each wave 64x32 (4x2 WMMA tiles).
// Double-buffered async global->LDS staging overlapped with WMMA compute.
// ---------------------------------------------------------------------------
#define GS 40   // LDS row stride (32 + 8 pad), keeps 16B alignment, spreads banks

__global__ __launch_bounds__(256)
void mha_gemm_bf16(const unsigned short* __restrict__ A,
                   const unsigned short* __restrict__ Wt,
                   unsigned short* __restrict__ outB,
                   float* __restrict__ outF,
                   int mode) {
#ifdef USE_ASYNC_LDS
  __shared__ alignas(16) unsigned short As[2][128 * GS];
  __shared__ alignas(16) unsigned short Bs[2][128 * GS];
#else
  __shared__ alignas(16) unsigned short As[1][128 * GS];
  __shared__ alignas(16) unsigned short Bs[1][128 * GS];
#endif

  const int lane = threadIdx.x & 31;
  const int wave = threadIdx.x >> 5;
  const int mBase = blockIdx.x * 128;
  const int nBase = blockIdx.y * 128;
  const int wm = wave >> 2;        // 0..1  -> 64-row half
  const int wn = wave & 3;         // 0..3  -> 32-col quarter
  const int nlane = lane & 15;
  const int half = lane >> 4;
  const int c0a = 8 * half;        // A-frag chunk base
  const int c0b = 8 * half;        // B-frag chunk base

  v8f c[4][2];
#pragma unroll
  for (int i = 0; i < 4; ++i)
#pragma unroll
    for (int j = 0; j < 2; ++j) c[i][j] = zero8();

#ifdef USE_ASYNC_LDS
  // Per-thread staging coordinates: chunks tid and tid+256 of 512 x 16B chunks
  // -> rows r0 and r0+64, same 8-element column offset.
  const int r0 = threadIdx.x >> 2;
  const int o0 = (threadIdx.x & 3) * 8;
  const unsigned short* gA0 = &A [(size_t)(mBase + r0)      * D_ + o0];
  const unsigned short* gA1 = &A [(size_t)(mBase + r0 + 64) * D_ + o0];
  const unsigned short* gB0 = &Wt[(size_t)(nBase + r0)      * D_ + o0];
  const unsigned short* gB1 = &Wt[(size_t)(nBase + r0 + 64) * D_ + o0];

  auto issue = [&](int kb, int stg) {
    ASYNC_CP_B128(gA0 + kb, &As[stg][(size_t)r0        * GS + o0]);
    ASYNC_CP_B128(gA1 + kb, &As[stg][(size_t)(r0 + 64) * GS + o0]);
    ASYNC_CP_B128(gB0 + kb, &Bs[stg][(size_t)r0        * GS + o0]);
    ASYNC_CP_B128(gB1 + kb, &Bs[stg][(size_t)(r0 + 64) * GS + o0]);
  };

  issue(0, 0);
  for (int ik = 0; ik < D_ / 32; ++ik) {
    ASYNC_WAIT0();          // my stage-ik tiles are in LDS
    __syncthreads();        // everyone's are; prior stage reads all finished
    if (ik + 1 < D_ / 32) issue((ik + 1) * 32, (ik + 1) & 1);  // overlap w/ WMMA

    const unsigned short* as = As[ik & 1];
    const unsigned short* bs = Bs[ik & 1];
    v16bf a[4], b[2];
#pragma unroll
    for (int i = 0; i < 4; ++i)
      a[i] = ldA(&as[(wm * 64 + 16 * i + nlane) * GS], c0a);
#pragma unroll
    for (int j = 0; j < 2; ++j)
      b[j] = ldB(&bs[(wn * 32 + 16 * j + nlane) * GS], c0b);
#pragma unroll
    for (int i = 0; i < 4; ++i)
#pragma unroll
      for (int j = 0; j < 2; ++j)
        c[i][j] = wmma_bf16(a[i], b[j], c[i][j]);
  }
#else
  for (int kb = 0; kb < D_; kb += 32) {
    __syncthreads();
#pragma unroll
    for (int c2 = threadIdx.x; c2 < 512; c2 += 256) {
      int row = c2 >> 2, off = (c2 & 3) * 8;
      *reinterpret_cast<uint4*>(&As[0][row * GS + off]) =
          *reinterpret_cast<const uint4*>(&A[(size_t)(mBase + row) * D_ + kb + off]);
      *reinterpret_cast<uint4*>(&Bs[0][row * GS + off]) =
          *reinterpret_cast<const uint4*>(&Wt[(size_t)(nBase + row) * D_ + kb + off]);
      if (kb + 32 < D_) {   // warm L2 for the next k-slab (global_prefetch_b8)
        __builtin_prefetch(&A[(size_t)(mBase + row) * D_ + kb + 32 + off], 0, 1);
        __builtin_prefetch(&Wt[(size_t)(nBase + row) * D_ + kb + 32 + off], 0, 1);
      }
    }
    __syncthreads();

    v16bf a[4], b[2];
#pragma unroll
    for (int i = 0; i < 4; ++i)
      a[i] = ldA(&As[0][(wm * 64 + 16 * i + nlane) * GS], c0a);
#pragma unroll
    for (int j = 0; j < 2; ++j)
      b[j] = ldB(&Bs[0][(wn * 32 + 16 * j + nlane) * GS], c0b);
#pragma unroll
    for (int i = 0; i < 4; ++i)
#pragma unroll
      for (int j = 0; j < 2; ++j)
        c[i][j] = wmma_bf16(a[i], b[j], c[i][j]);
  }
#endif

  // epilogue: C layout => row M = r + 8*half, col N = lane&15 (per 16x16 tile)
#pragma unroll
  for (int i = 0; i < 4; ++i)
#pragma unroll
    for (int j = 0; j < 2; ++j)
#pragma unroll
      for (int r = 0; r < 8; ++r) {
        int m = mBase + wm * 64 + 16 * i + r + 8 * half;
        int n = nBase + wn * 32 + 16 * j + nlane;
        float val = c[i][j][r];
        if (mode == 0) {
          int h = n >> 6, dh = n & 63;
          int bb = m >> 11, t = m & (T_ - 1);
          outB[((size_t)((bb * H_ + h) * T_ + t)) * DH_ + dh] = f2bf(val);
        } else {
          outF[(size_t)m * D_ + n] = val;
        }
      }
}

// ---------------------------------------------------------------------------
// Kernel 3: flash attention per (b, h, 128-row q tile)
//   Q,K,V head-major bf16 [B,H,T,Dh]; output bf16 [BT, D] (head-merged)
// K/V tiles are small (256 KB/head) and L2-resident on a 192 MB L2, so the
// staging here stays synchronous; we just prefetch the next key tile.
// ---------------------------------------------------------------------------
#define QS 72   // 64 + 8 pad; 72*2 bytes = 144, multiple of 16

__global__ __launch_bounds__(256)
void mha_flash_attn(const unsigned short* __restrict__ Qp,
                    const unsigned short* __restrict__ Kp,
                    const unsigned short* __restrict__ Vp,
                    const unsigned char* __restrict__ mask,
                    unsigned short* __restrict__ attnB) {
  __shared__ alignas(16) unsigned short Qs[128 * QS];
  __shared__ alignas(16) unsigned short Ks[64 * QS];
  __shared__ alignas(16) unsigned short Vt[64 * QS];        // transposed: [dh][key]
  __shared__ alignas(16) unsigned short Ps[8][16 * QS];     // per-wave P tile

  const int lane = threadIdx.x & 31;
  const int wave = threadIdx.x >> 5;
  const int qBase = blockIdx.x * 128;
  const int h = blockIdx.y;
  const int bb = blockIdx.z;
  const size_t headOff = ((size_t)(bb * H_ + h)) * T_ * DH_;
  const unsigned char* mrow = mask + (size_t)bb * T_ * T_;

  const int nlane = lane & 15;
  const int half = lane >> 4;
  const int c0a = 8 * half;
  const int c0b = 8 * half;
  const int qw = qBase + wave * 16;

  // stage Q tile [128 x 64]
#pragma unroll
  for (int c2 = threadIdx.x; c2 < 1024; c2 += 256) {
    int row = c2 >> 3, off = (c2 & 7) * 8;
    *reinterpret_cast<uint4*>(&Qs[row * QS + off]) =
        *reinterpret_cast<const uint4*>(&Qp[headOff + (size_t)(qBase + row) * DH_ + off]);
  }

  float mi[8], li[8];
  v8f o[4];
#pragma unroll
  for (int r = 0; r < 8; ++r) { mi[r] = -INFINITY; li[r] = 0.f; }
#pragma unroll
  for (int nt = 0; nt < 4; ++nt) o[nt] = zero8();

  for (int kt = 0; kt < T_; kt += 64) {
    __syncthreads();
    // stage K tile [64 x 64]
#pragma unroll
    for (int c2 = threadIdx.x; c2 < 512; c2 += 256) {
      int row = c2 >> 3, off = (c2 & 7) * 8;
      *reinterpret_cast<uint4*>(&Ks[row * QS + off]) =
          *reinterpret_cast<const uint4*>(&Kp[headOff + (size_t)(kt + row) * DH_ + off]);
      if (kt + 64 < T_) {
        __builtin_prefetch(&Kp[headOff + (size_t)(kt + 64 + row) * DH_ + off], 0, 1);
        __builtin_prefetch(&Vp[headOff + (size_t)(kt + 64 + row) * DH_ + off], 0, 1);
      }
    }
    // stage V tile transposed -> Vt[dh][key]
    {
      int row = threadIdx.x & 63;
      int colBase = (threadIdx.x >> 6) * 16;
      FragCvt uu;
      uu.u[0] = *reinterpret_cast<const uint4*>(&Vp[headOff + (size_t)(kt + row) * DH_ + colBase]);
      uu.u[1] = *reinterpret_cast<const uint4*>(&Vp[headOff + (size_t)(kt + row) * DH_ + colBase + 8]);
#pragma unroll
      for (int e = 0; e < 16; ++e) Vt[(colBase + e) * QS + row] = uu.s[e];
    }
    __syncthreads();

    // S = Q * K^T  (reduce over dh=64 -> 2 WMMA k-steps)
    v8f s[4];
#pragma unroll
    for (int f = 0; f < 4; ++f) s[f] = zero8();
#pragma unroll
    for (int ks = 0; ks < 2; ++ks) {
      v16bf aq = ldA(&Qs[(wave * 16 + nlane) * QS], ks * 32 + c0a);
#pragma unroll
      for (int f = 0; f < 4; ++f) {
        v16bf bk = ldB(&Ks[(16 * f + nlane) * QS], ks * 32 + c0b);
        s[f] = wmma_bf16(aq, bk, s[f]);
      }
    }

    // scale, mask, online softmax (rows r / r+8 of this wave's 16-row strip)
#pragma unroll
    for (int r = 0; r < 8; ++r) {
      int qg = qw + r + 8 * half;
      float sv[4];
      float rm = -INFINITY;
#pragma unroll
      for (int f = 0; f < 4; ++f) {
        float v = s[f][r] * 0.125f;                 // 1/sqrt(Dh)
        int kg = kt + 16 * f + nlane;
        if (mrow[(size_t)qg * T_ + kg]) v = -INFINITY;
        sv[f] = v;
        rm = fmaxf(rm, v);
      }
      rm = redmax16(rm);
      float mn = fmaxf(mi[r], rm);
      float alpha = (mn == -INFINITY) ? 0.f : __expf(mi[r] - mn);
      float rs = 0.f;
#pragma unroll
      for (int f = 0; f < 4; ++f) {
        float p = __expf(sv[f] - mn);
        rs += p;
        Ps[wave][(r + 8 * half) * QS + 16 * f + nlane] = f2bf(p);
      }
      rs = redsum16(rs);
      li[r] = li[r] * alpha + rs;
      mi[r] = mn;
#pragma unroll
      for (int nt = 0; nt < 4; ++nt) o[nt][r] *= alpha;
    }

    // O += P * V   (reduce over 64 keys -> 2 WMMA k-steps, Vt rows contiguous in key)
#pragma unroll
    for (int ks = 0; ks < 2; ++ks) {
      v16bf pa = ldA(&Ps[wave][nlane * QS], ks * 32 + c0a);
#pragma unroll
      for (int nt = 0; nt < 4; ++nt) {
        v16bf vb = ldB(&Vt[(16 * nt + nlane) * QS], ks * 32 + c0b);
        o[nt] = wmma_bf16(pa, vb, o[nt]);
      }
    }
  }

  // epilogue: normalize by l and store head-merged bf16 [BT, D]
#pragma unroll
  for (int nt = 0; nt < 4; ++nt)
#pragma unroll
    for (int r = 0; r < 8; ++r) {
      int qg = qw + r + 8 * half;
      float val = o[nt][r] / li[r];
      attnB[(size_t)(bb * T_ + qg) * D_ + h * DH_ + 16 * nt + nlane] = f2bf(val);
    }
}

// ---------------------------------------------------------------------------
// Host-side launch
// ---------------------------------------------------------------------------
extern "C" void kernel_launch(void* const* d_in, const int* in_sizes, int n_in,
                              void* d_out, int out_size, void* d_ws, size_t ws_size,
                              hipStream_t stream) {
  const float* q  = (const float*)d_in[0];
  const float* k  = (const float*)d_in[1];
  const float* v  = (const float*)d_in[2];
  const unsigned char* mask = (const unsigned char*)d_in[3];
  const float* Wq = (const float*)d_in[4];
  const float* Wk = (const float*)d_in[5];
  const float* Wv = (const float*)d_in[6];
  const float* Wo = (const float*)d_in[7];
  float* out = (float*)d_out;

  const size_t NE = (size_t)BT_ * D_;   // 4,194,304 elements
  const size_t WE = (size_t)D_ * D_;    // 1,048,576 elements

  unsigned short* ws  = (unsigned short*)d_ws;
  unsigned short* qbf = ws;
  unsigned short* kbf = qbf + NE;
  unsigned short* vbf = kbf + NE;
  unsigned short* wqb = vbf + NE;
  unsigned short* wkb = wqb + WE;
  unsigned short* wvb = wkb + WE;
  unsigned short* wob = wvb + WE;
  unsigned short* Qp  = wob + WE;
  unsigned short* Kp  = Qp + NE;
  unsigned short* Vp  = Kp + NE;
  unsigned short* attnb = Vp + NE;      // total 64 MB of bf16 scratch

  // 1) f32 -> bf16 conversions
  mha_cvt_bf16<<<(int)(NE / 4 / 256), 256, 0, stream>>>(q,  qbf, (int)NE);
  mha_cvt_bf16<<<(int)(NE / 4 / 256), 256, 0, stream>>>(k,  kbf, (int)NE);
  mha_cvt_bf16<<<(int)(NE / 4 / 256), 256, 0, stream>>>(v,  vbf, (int)NE);
  mha_cvt_bf16<<<(int)(WE / 4 / 256), 256, 0, stream>>>(Wq, wqb, (int)WE);
  mha_cvt_bf16<<<(int)(WE / 4 / 256), 256, 0, stream>>>(Wk, wkb, (int)WE);
  mha_cvt_bf16<<<(int)(WE / 4 / 256), 256, 0, stream>>>(Wv, wvb, (int)WE);
  mha_cvt_bf16<<<(int)(WE / 4 / 256), 256, 0, stream>>>(Wo, wob, (int)WE);

  // 2) Q/K/V projections (WMMA GEMM, head-major scatter)
  dim3 ggrid(BT_ / 128, D_ / 128);
  mha_gemm_bf16<<<ggrid, 256, 0, stream>>>(qbf, wqb, Qp, nullptr, 0);
  mha_gemm_bf16<<<ggrid, 256, 0, stream>>>(kbf, wkb, Kp, nullptr, 0);
  mha_gemm_bf16<<<ggrid, 256, 0, stream>>>(vbf, wvb, Vp, nullptr, 0);

  // 3) flash attention
  dim3 agrid(T_ / 128, H_, B_);
  mha_flash_attn<<<agrid, 256, 0, stream>>>(Qp, Kp, Vp, mask, attnb);

  // 4) output projection -> f32 d_out
  mha_gemm_bf16<<<ggrid, 256, 0, stream>>>(attnb, wob, nullptr, out, 1);
}
